// RetrievalModule_38371237823154
// MI455X (gfx1250) — matrix operand order repaired and checked by reading, not compiled, not run on hardware
//
#include <hip/hip_runtime.h>
#include <math.h>

typedef __attribute__((ext_vector_type(16))) __bf16 v16bf;
typedef __attribute__((ext_vector_type(8)))  __bf16 v8bf;
typedef __attribute__((ext_vector_type(8)))  float  v8f;

#define NEG_INF (-__builtin_inff())
#define D 768
#define B 64
#define TOPK 10
#define CHUNK 128   // db rows per block-iteration (8 waves x 16 rows)
#define KSTEPS (D / 32)

// ---------------------------------------------------------------------------
// Kernel 1: L2-normalize the 64 queries, store as bf16 (row-major [64][768]).
// ---------------------------------------------------------------------------
__global__ void __launch_bounds__(256)
qnorm_kernel(const float* __restrict__ q, unsigned short* __restrict__ qbf_raw) {
  __bf16* qbf = reinterpret_cast<__bf16*>(qbf_raw);
  const int row  = blockIdx.x;
  const int tid  = threadIdx.x;
  const int wave = tid >> 5, lane = tid & 31;
  const float* qr = q + row * D;
  float vals[3];
  float s = 0.f;
#pragma unroll
  for (int i = 0; i < 3; ++i) { vals[i] = qr[tid + i * 256]; s += vals[i] * vals[i]; }
#pragma unroll
  for (int off = 16; off; off >>= 1) s += __shfl_xor(s, off, 32);
  __shared__ float wsum[8];
  if (lane == 0) wsum[wave] = s;
  __syncthreads();
  float tot = 0.f;
#pragma unroll
  for (int w = 0; w < 8; ++w) tot += wsum[w];
  const float inv = 1.f / fmaxf(sqrtf(tot), 1e-12f);
  __bf16* o = qbf + row * D;
#pragma unroll
  for (int i = 0; i < 3; ++i) o[tid + i * 256] = (__bf16)(vals[i] * inv);
}

// ---------------------------------------------------------------------------
// Async HBM -> LDS staging: 4 x b128 per lane (64 B = this lane's half-row
// K-slice for one k-step). Tracked by ASYNCcnt, throttled with s_wait_asynccnt.
// ---------------------------------------------------------------------------
__device__ __forceinline__ void issue_stage(unsigned ldsoff, const float* gsrc) {
  unsigned long long g = (unsigned long long)(uintptr_t)gsrc;
#pragma unroll
  for (int i = 0; i < 4; ++i)
    asm volatile("global_load_async_to_lds_b128 %0, %1, off"
                 :: "v"(ldsoff + (unsigned)(i * 16)), "v"(g + (unsigned long long)(i * 16))
                 : "memory");
}

// ---------------------------------------------------------------------------
// Kernel 2: streaming bf16-WMMA GEMM (64 x 768 x N) fused with db-row norm and
// per-block running top-10 per query.
//   Wave w of 8 owns db rows [chunk*128 + w*16, +16) x all 64 queries.
//   db stream is double-buffered through LDS via GLOBAL_LOAD_ASYNC_TO_LDS_B128
//   (2 k-steps in flight); the staging buffers overlay the wave's own slice of
//   the sims matrix (live ranges disjoint, fenced by the chunk barrier).
// ---------------------------------------------------------------------------
__global__ void __launch_bounds__(256)
sim_topk_kernel(const unsigned short* __restrict__ qbf_raw,
                const float* __restrict__ db,
                float* __restrict__ candS, int* __restrict__ candI,
                int N, int nChunks) {
  const __bf16* qbf = reinterpret_cast<const __bf16*>(qbf_raw);
  __shared__ float sims[CHUNK * B];     // 32 KB; doubles as async staging
  __shared__ float topS[B * TOPK];      // running top-10 scores (sorted desc)
  __shared__ int   topI[B * TOPK];

  const int tid  = threadIdx.x;
  const int wave = tid >> 5;
  const int lane = tid & 31;
  const int lo   = lane & 15;   // db-row within tile / query-row within A tile
  const int hi   = lane >> 4;   // lane-half selector for K ranges

  // This lane's staging slot (LDS byte offset = low 32 bits of flat address).
  // Wave region = wave*4096 B (== the wave's 16 sims rows); two 2 KB buffers.
  const unsigned ldsLane =
      (unsigned)(uintptr_t)(&sims[0]) + (unsigned)(wave * 4096 + lo * 128 + hi * 64);
  const int fidxLane = wave * 1024 + (lo * 2 + hi) * 16;   // float index of slot

  for (int i = tid; i < B * TOPK; i += 256) { topS[i] = NEG_INF; topI[i] = -1; }
  __syncthreads();

  for (int chunk = blockIdx.x; chunk < nChunks; chunk += gridDim.x) {
    const int base = chunk * CHUNK;
    const int row  = base + wave * 16 + lo;
    const int rowc = (row < N) ? row : (N - 1);
    const float* dbrow = db + (size_t)rowc * D + hi * 16;

    { // L2 prefetch of the rows this lane will stream next chunk
      int nrow = row + gridDim.x * CHUNK;
      if (nrow < N) __builtin_prefetch(db + (size_t)nrow * D + hi * 16, 0, 1);
    }

    v8f acc[4];
    const v8f zero = {0.f, 0.f, 0.f, 0.f, 0.f, 0.f, 0.f, 0.f};
#pragma unroll
    for (int mt = 0; mt < 4; ++mt) acc[mt] = zero;
    float sumsq = 0.f;

    // prologue: fill both staging buffers (k-steps 0 and 1)
    issue_stage(ldsLane,          dbrow + 0);
    issue_stage(ldsLane + 2048u,  dbrow + 32);

    for (int step = 0; step < KSTEPS; ++step) {
      const int k0 = step * 32;
      const unsigned bufB = ldsLane + (unsigned)((step & 1) * 2048);
      // current buffer complete; next buffer's 4 loads may still be in flight
      if (step + 1 < KSTEPS) asm volatile("s_wait_asynccnt 4" ::: "memory");
      else                   asm volatile("s_wait_asynccnt 0" ::: "memory");

      // ---- B fragment from LDS: 32x16 K x N bf16. Lane = column n (db row);
      //      halves 0..15 = db[n][k0 + hi*16 + 0..15].
      float f[16];
      {
        const float4* lp =
            reinterpret_cast<const float4*>(&sims[fidxLane + (step & 1) * 512]);
        float4 t;
        t = lp[0]; f[0]  = t.x; f[1]  = t.y; f[2]  = t.z; f[3]  = t.w;
        t = lp[1]; f[4]  = t.x; f[5]  = t.y; f[6]  = t.z; f[7]  = t.w;
        t = lp[2]; f[8]  = t.x; f[9]  = t.y; f[10] = t.z; f[11] = t.w;
        t = lp[3]; f[12] = t.x; f[13] = t.y; f[14] = t.z; f[15] = t.w;
      }
      v16bf bfrag;
#pragma unroll
      for (int e = 0; e < 16; ++e) { sumsq += f[e] * f[e]; bfrag[e] = (__bf16)f[e]; }

      // ---- A fragments: 16x32 M x K bf16 per query tile (L2-resident). Lane
      //      = row m; halves 0..15 = q[m][k0+hi*16+0..15].
#pragma unroll
      for (int mt = 0; mt < 4; ++mt) {
        const __bf16* qr = qbf + (size_t)(mt * 16 + lo) * D + k0 + hi * 16;
        v8bf a0 = *reinterpret_cast<const v8bf*>(qr);
        v8bf a1 = *reinterpret_cast<const v8bf*>(qr + 8);
        v16bf afrag;
#pragma unroll
        for (int e = 0; e < 8; ++e) { afrag[e] = a0[e]; afrag[8 + e] = a1[e]; }
        acc[mt] = __builtin_amdgcn_wmma_f32_16x16x32_bf16(
            false, afrag, false, bfrag, (short)0, acc[mt], false, false);
      }

      // refill the buffer we just drained with k-step step+2.
      // ISA 08_async_tensor §4.1: async LDS writes are NOT ordered with DS
      // reads, so drain DScnt before overwriting the drained buffer (nearly
      // free: the cvts above already force DScnt -> 0).
      if (step + 2 < KSTEPS) {
        asm volatile("s_wait_dscnt 0" ::: "memory");
        issue_stage(bufB, dbrow + (k0 + 64));
      }
    }

    // db row norm: this lane covered half of row 'rowc'; partner lane (xor 16)
    // covered the other half. D layout: lane's N column == its own db row.
    sumsq += __shfl_xor(sumsq, 16, 32);
    const float inv   = 1.f / fmaxf(sqrtf(sumsq), 1e-12f);
    const int   nloc  = wave * 16 + lo;
    const bool  valid = (row < N);
#pragma unroll
    for (int mt = 0; mt < 4; ++mt)
#pragma unroll
      for (int v = 0; v < 8; ++v) {
        const int qq = mt * 16 + hi * 8 + v;       // M = v + 8*hi in tile mt
        sims[nloc * B + qq] = valid ? acc[mt][v] * inv : NEG_INF;
      }
    __syncthreads();

    // ---- running top-10 merge: wave owns queries [wave*8, wave*8+8)
    for (int j = 0; j < 8; ++j) {
      const int q = wave * 8 + j;
      float c0 = sims[(lane      ) * B + q];
      float c1 = sims[(lane + 32 ) * B + q];
      float c2 = sims[(lane + 64 ) * B + q];
      float c3 = sims[(lane + 96 ) * B + q];
      const int i0 = base + lane, i1 = base + lane + 32,
                i2 = base + lane + 64, i3 = base + lane + 96;
      // prefilter: skip chunk if nothing beats current 10th best
      float m = fmaxf(fmaxf(c0, c1), fmaxf(c2, c3));
#pragma unroll
      for (int off = 16; off; off >>= 1) m = fmaxf(m, __shfl_xor(m, off, 32));
      if (m <= topS[q * TOPK + 9]) continue;

      float os = (lane < TOPK) ? topS[q * TOPK + lane] : NEG_INF;
      int   oi = (lane < TOPK) ? topI[q * TOPK + lane] : -1;
      float newS = NEG_INF; int newI = -1;
      for (int i = 0; i < TOPK; ++i) {
        float bs = os; int bi = oi;
        if (c0 > bs || (c0 == bs && i0 < bi)) { bs = c0; bi = i0; }
        if (c1 > bs || (c1 == bs && i1 < bi)) { bs = c1; bi = i1; }
        if (c2 > bs || (c2 == bs && i2 < bi)) { bs = c2; bi = i2; }
        if (c3 > bs || (c3 == bs && i3 < bi)) { bs = c3; bi = i3; }
#pragma unroll
        for (int off = 16; off; off >>= 1) {
          float s2 = __shfl_xor(bs, off, 32);
          int   b2 = __shfl_xor(bi, off, 32);
          if (s2 > bs || (s2 == bs && b2 < bi)) { bs = s2; bi = b2; }
        }
        if (lane == i) { newS = bs; newI = bi; }
        if (i0 == bi) c0 = NEG_INF;
        if (i1 == bi) c1 = NEG_INF;
        if (i2 == bi) c2 = NEG_INF;
        if (i3 == bi) c3 = NEG_INF;
        if (oi == bi) os = NEG_INF;
      }
      if (lane < TOPK) { topS[q * TOPK + lane] = newS; topI[q * TOPK + lane] = newI; }
    }
    __syncthreads();   // protect sims/staging before next chunk overwrites
  }

  for (int i = tid; i < B * TOPK; i += 256) {
    candS[(size_t)blockIdx.x * (B * TOPK) + i] = topS[i];
    candI[(size_t)blockIdx.x * (B * TOPK) + i] = topI[i];
  }
}

// ---------------------------------------------------------------------------
// Kernel 3: reduce NB per-block candidate lists to the global top-10 per query.
// One block per query; 10 rounds of block-wide argmax with exclusion by index.
// Output layout: scores [64][10] then indices-as-float [64][10].
// ---------------------------------------------------------------------------
__global__ void __launch_bounds__(256)
final_topk_kernel(const float* __restrict__ candS, const int* __restrict__ candI,
                  float* __restrict__ out, int NB) {
  const int q    = blockIdx.x;
  const int tid  = threadIdx.x;
  const int wave = tid >> 5, lane = tid & 31;
  const int total = NB * TOPK;
  __shared__ float redS[8];
  __shared__ int   redI[8];
  __shared__ float bcS;
  __shared__ int   bcI;
  int sel[TOPK];

  for (int i = 0; i < TOPK; ++i) {
    float bs = NEG_INF; int bi = -1;
    for (int c = tid; c < total; c += 256) {
      const int b = c / TOPK, s = c % TOPK;
      const size_t off = (size_t)b * (B * TOPK) + q * TOPK + s;
      const float sc = candS[off];
      const int   id = candI[off];
      bool taken = false;
      for (int j = 0; j < i; ++j) taken = taken || (id == sel[j]);
      if (!taken && (sc > bs || (sc == bs && (unsigned)id < (unsigned)bi))) { bs = sc; bi = id; }
    }
#pragma unroll
    for (int off = 16; off; off >>= 1) {
      float s2 = __shfl_down(bs, off, 32);
      int   b2 = __shfl_down(bi, off, 32);
      if (s2 > bs || (s2 == bs && (unsigned)b2 < (unsigned)bi)) { bs = s2; bi = b2; }
    }
    if (lane == 0) { redS[wave] = bs; redI[wave] = bi; }
    __syncthreads();
    if (wave == 0) {
      bs = (lane < 8) ? redS[lane] : NEG_INF;
      bi = (lane < 8) ? redI[lane] : -1;
#pragma unroll
      for (int off = 4; off; off >>= 1) {
        float s2 = __shfl_down(bs, off, 32);
        int   b2 = __shfl_down(bi, off, 32);
        if (s2 > bs || (s2 == bs && (unsigned)b2 < (unsigned)bi)) { bs = s2; bi = b2; }
      }
      if (lane == 0) { bcS = bs; bcI = bi; }
    }
    __syncthreads();
    sel[i] = bcI;
    if (tid == 0) {
      out[q * TOPK + i]            = bcS;
      out[B * TOPK + q * TOPK + i] = (float)bcI;
    }
    __syncthreads();
  }
}

// ---------------------------------------------------------------------------
extern "C" void kernel_launch(void* const* d_in, const int* in_sizes, int n_in,
                              void* d_out, int out_size, void* d_ws, size_t ws_size,
                              hipStream_t stream) {
  const float* query = (const float*)d_in[0];
  const float* db    = (const float*)d_in[1];
  const int N       = in_sizes[1] / D;
  const int nChunks = (N + CHUNK - 1) / CHUNK;

  char* ws = (char*)d_ws;
  const size_t qbfBytes = (size_t)B * D * 2;            // 96 KB bf16 queries
  const size_t candOff  = (qbfBytes + 255) & ~(size_t)255;
  const size_t perBlock = (size_t)(B * TOPK) * (sizeof(float) + sizeof(int));
  size_t avail = (ws_size > candOff) ? (ws_size - candOff) : 0;
  int NB = (int)(avail / perBlock);
  if (NB > 1024)    NB = 1024;
  if (NB > nChunks) NB = nChunks;
  if (NB < 1)       NB = 1;

  unsigned short* qbf = (unsigned short*)ws;
  float* candS = (float*)(ws + candOff);
  int*   candI = (int*)(ws + candOff + (size_t)NB * (B * TOPK) * sizeof(float));

  qnorm_kernel<<<B, 256, 0, stream>>>(query, qbf);
  sim_topk_kernel<<<NB, 256, 0, stream>>>(qbf, db, candS, candI, N, nChunks);
  final_topk_kernel<<<B, 256, 0, stream>>>(candS, candI, (float*)d_out, NB);
}
